// DoubleQValueNet_22058952032357
// MI455X (gfx1250) — matrix operand order repaired and checked by reading, not compiled, not run on hardware
//
#include <hip/hip_runtime.h>
#include <hip/hip_bf16.h>
#include <stdint.h>

// ---------------- problem constants (from reference) ----------------
#define FDIM   64
#define HLDIM  512
#define NEDGES 200000
#define K0PAD  160        // 2F+4 = 132 padded to a multiple of 32
#define NSG0   16384
#define NSG1   8192
#define GROWS  131072     // both subgraph gathers produce 131072 rows

typedef __bf16 bf16;
typedef __attribute__((ext_vector_type(16))) __bf16 v16bf;
typedef __attribute__((ext_vector_type(8)))  float  v8f;

__device__ __forceinline__ float lrelu(float v) { return v > 0.f ? v : 0.01f * v; }

__device__ __forceinline__ v8f wmma_bf16(v16bf a, v16bf b, v8f c) {
  // D = A(16x32 bf16) * B(32x16 bf16) + C(16x16 f32)
  return __builtin_amdgcn_wmma_f32_16x16x32_bf16(false, a, false, b, (short)0, c, false, false);
}

// ---------------- CDNA5 async global->LDS copy (ASYNCcnt-tracked) ----------------
// global_load_async_to_lds_b128: VDST = LDS byte address (low 32 bits of a
// generic pointer into the LDS aperture), VADDR = 64-bit global address, SADDR=off.
__device__ __forceinline__ void async_copy_b128(void* lds_dst, const void* gsrc) {
  uint32_t l = (uint32_t)(uintptr_t)lds_dst;
  uint64_t g = (uint64_t)(uintptr_t)gsrc;
  asm volatile("global_load_async_to_lds_b128 %0, %1, off" :: "v"(l), "v"(g) : "memory");
}
__device__ __forceinline__ void async_wait0() {
#if __has_builtin(__builtin_amdgcn_s_wait_asynccnt)
  __builtin_amdgcn_s_wait_asynccnt(0);
#else
  asm volatile("s_wait_asynccnt 0" ::: "memory");
#endif
}

// ------------- fragment loaders (CDNA5 ISA 7.12.2 layouts, wave32) -------------

// A-matrix 16x32 fragment from row-major [M][K] bf16.
// Lanes 0-15: M=lane, K = {0..7,16..23}; lanes 16-31: M=lane-16, K = {8..15,24..31}.
__device__ __forceinline__ v16bf load_a(const bf16* mat, int ld, int mBase, int kBase) {
  int lane = threadIdx.x & 31;
  const bf16* p = mat + (size_t)(mBase + (lane & 15)) * ld + kBase + ((lane & 16) ? 8 : 0);
  v16bf f;
#pragma unroll
  for (int r = 0; r < 8; ++r) {
    int k = ((r & 4) << 2) + 2 * (r & 3);   // 0,2,4,6,16,18,20,22
    f[2 * r]     = p[k];
    f[2 * r + 1] = p[k + 1];
  }
  return f;
}

// Same A layout but sourced from f32 (converted to bf16 on load).
__device__ __forceinline__ v16bf load_a_f32(const float* mat, int ld, int mBase, int kBase) {
  int lane = threadIdx.x & 31;
  const float* p = mat + (size_t)(mBase + (lane & 15)) * ld + kBase + ((lane & 16) ? 8 : 0);
  v16bf f;
#pragma unroll
  for (int r = 0; r < 8; ++r) {
    int k = ((r & 4) << 2) + 2 * (r & 3);
    f[2 * r]     = (bf16)p[k];
    f[2 * r + 1] = (bf16)p[k + 1];
  }
  return f;
}

// A fragment with the GGNN ring aggregation fused in: row m reads its two ring
// neighbours inside its subgraph (power-of-two size, mask = sg-1) and sums them.
__device__ __forceinline__ v16bf load_a_ringsum(const bf16* mat, int ld, int mBase,
                                                int kBase, int sgMask) {
  int lane = threadIdx.x & 31;
  int row = mBase + (lane & 15);
  int hi = row & ~sgMask, j = row & sgMask;
  int ko = kBase + ((lane & 16) ? 8 : 0);
  const bf16* pp = mat + (size_t)(hi | ((j + sgMask) & sgMask)) * ld + ko;
  const bf16* pn = mat + (size_t)(hi | ((j + 1) & sgMask)) * ld + ko;
  v16bf f;
#pragma unroll
  for (int r = 0; r < 8; ++r) {
    int k = ((r & 4) << 2) + 2 * (r & 3);
    f[2 * r]     = (bf16)((float)pp[k]     + (float)pn[k]);
    f[2 * r + 1] = (bf16)((float)pp[k + 1] + (float)pn[k + 1]);
  }
  return f;
}

// B-matrix 32x16 fragment. Weights stored transposed [N][K] row-major; K order
// matches A (lanes 0-15: K {0..7,16..23}, lanes 16-31: K {8..15,24..31}).
__device__ __forceinline__ v16bf load_b(const bf16* wt, int ld, int nBase, int kBase) {
  int lane = threadIdx.x & 31;
  const bf16* p = wt + (size_t)(nBase + (lane & 15)) * ld + kBase + ((lane & 16) ? 8 : 0);
  v16bf f;
#pragma unroll
  for (int e = 0; e < 16; ++e) f[e] = p[(e & 7) + ((e & 8) << 1)];
  return f;
}

// Store a 16x16 f32 C tile (8 VGPRs, ISA layout) to bf16 with bias (+ optional LReLU).
__device__ __forceinline__ void store_c_bf16(bf16* dst, int ld, int mBase, int nBase,
                                             v8f c, const float* bias, bool act) {
  int lane = threadIdx.x & 31;
  int col = nBase + (lane & 15);
  int rOff = (lane & 16) ? 8 : 0;
  float b = bias ? bias[col] : 0.f;
#pragma unroll
  for (int r = 0; r < 8; ++r) {
    float v = c[r] + b;
    if (act) v = lrelu(v);
    dst[(size_t)(mBase + r + rOff) * ld + col] = (bf16)v;
  }
}

// ---- 2M x 4N register blocking: each B fragment feeds 2 WMMAs, each A feeds 4 ----
// One wave computes a [32 x 64] C panel (8 accumulators). Per K-step:
// 2 A-frag LDS reads + 4 B-frag L2 reads -> 8 WMMAs (halves weight traffic vs 1M).
struct CPanel2 { v8f c[8]; };   // c[q] = (m=0, n=q*16); c[4+q] = (m=16, n=q*16)
__device__ __forceinline__ CPanel2 gemm_panel2(const bf16* A, int lda,
                                               const bf16* Wt, int K, int n0) {
  CPanel2 p;
#pragma unroll
  for (int i = 0; i < 8; ++i) p.c[i] = (v8f){};
  for (int kb = 0; kb < K; kb += 32) {
    v16bf a0 = load_a(A, lda, 0, kb);
    v16bf a1 = load_a(A, lda, 16, kb);
#pragma unroll
    for (int q = 0; q < 4; ++q) {
      v16bf b = load_b(Wt, K, n0 + 16 * q, kb);
      p.c[q]     = wmma_bf16(a0, b, p.c[q]);
      p.c[4 + q] = wmma_bf16(a1, b, p.c[4 + q]);
    }
  }
  return p;
}

// ================= kernel: weight convert + transpose (f32 [K][N] -> bf16 [N][Kpad]) ==
__global__ __launch_bounds__(256) void wprep_kernel(const float* __restrict__ w, int K,
                                                    int Nc, int Kpad, bf16* __restrict__ wt) {
  int i = blockIdx.x * 256 + threadIdx.x;
  if (i >= Nc * Kpad) return;
  int n = i / Kpad, k = i - n * Kpad;
  wt[i] = (bf16)(k < K ? w[(size_t)k * Nc + n] : 0.f);
}

// ================= kernel: fused 3-layer edge MLP (QGNN) ============================
// 32-edge tile per workgroup; all activations live in 64KB LDS; bf16 WMMA, f32 accum.
// Each of the 8 waves owns one 64-column output panel across all 32 rows.
__global__ __launch_bounds__(256) void qgnn_kernel(
    const float* __restrict__ x, const float* __restrict__ actions,
    const float* __restrict__ angles, const int* __restrict__ ei,
    const bf16* __restrict__ w0t, const float* __restrict__ b0,
    const bf16* __restrict__ w1t, const float* __restrict__ b1,
    const bf16* __restrict__ w2t, const float* __restrict__ b2,
    bf16* __restrict__ ef) {
  __shared__ bf16 bufA[32 * HLDIM];   // 32KB
  __shared__ bf16 bufB[32 * HLDIM];   // 32KB (input tile lives in first 32*160)
  const int e0 = blockIdx.x * 32;
  const int tid = threadIdx.x;
  const int wave = tid >> 5;

  // gather x[src] || x[dst] || [action, angles] into [32][160] (zero padded)
  for (int i = tid; i < 32 * K0PAD; i += 256) {
    int m = i / K0PAD, k = i - m * K0PAD;
    int e = e0 + m;
    float v = 0.f;
    if (e < NEDGES) {
      if (k < FDIM)              v = x[(size_t)ei[e] * FDIM + k];
      else if (k < 2 * FDIM)     v = x[(size_t)ei[NEDGES + e] * FDIM + (k - FDIM)];
      else if (k == 2 * FDIM)    v = actions[e];
      else if (k < 2 * FDIM + 4) v = angles[(size_t)e * 3 + (k - 2 * FDIM - 1)];
    }
    bufB[m * K0PAD + k] = (bf16)v;
  }
  __syncthreads();

  // layer 0: [32,160] x [160,512] -> bufA, LReLU (one 64-col panel per wave)
  {
    int n0 = wave * 64;
    CPanel2 p = gemm_panel2(bufB, K0PAD, w0t, K0PAD, n0);
#pragma unroll
    for (int q = 0; q < 4; ++q) {
      store_c_bf16(bufA, HLDIM, 0,  n0 + 16 * q, p.c[q],     b0, true);
      store_c_bf16(bufA, HLDIM, 16, n0 + 16 * q, p.c[4 + q], b0, true);
    }
  }
  __syncthreads();

  // layer 1: [32,512] x [512,512] -> bufB, LReLU
  {
    int n0 = wave * 64;
    CPanel2 p = gemm_panel2(bufA, HLDIM, w1t, HLDIM, n0);
#pragma unroll
    for (int q = 0; q < 4; ++q) {
      store_c_bf16(bufB, HLDIM, 0,  n0 + 16 * q, p.c[q],     b1, true);
      store_c_bf16(bufB, HLDIM, 16, n0 + 16 * q, p.c[4 + q], b1, true);
    }
  }
  __syncthreads();

  // layer 2: [32,512] x [512,64] -> global ef (bf16), bias only (8 tiles, 1/wave)
  {
    int mt = (wave & 1) * 16, nt = (wave >> 1) * 16;
    v8f c = {};
    for (int kb = 0; kb < HLDIM; kb += 32)
      c = wmma_bf16(load_a(bufB, HLDIM, mt, kb), load_b(w2t, HLDIM, nt, kb), c);
    int lane = tid & 31;
    int col = nt + (lane & 15);
    int rOff = (lane & 16) ? 8 : 0;
    float bb = b2[col];
#pragma unroll
    for (int r = 0; r < 8; ++r) {
      int e = e0 + mt + r + rOff;
      if (e < NEDGES) ef[(size_t)e * FDIM + col] = (bf16)(c[r] + bb);
    }
  }
}

// ================= kernel: side loss (sigmoid MSE), pre-scaled atomics =============
__global__ __launch_bounds__(256) void side_kernel(const bf16* __restrict__ ef,
                                                   const float* __restrict__ ws,
                                                   const float* __restrict__ bs,
                                                   const float* __restrict__ gt,
                                                   float scale, float* __restrict__ out) {
  __shared__ float wsh[FDIM];
  __shared__ float red[256];
  if (threadIdx.x < FDIM) wsh[threadIdx.x] = ws[threadIdx.x];
  __syncthreads();
  int e = blockIdx.x * 256 + threadIdx.x;
  float err2 = 0.f;
  if (e < NEDGES) {
    float acc = bs[0];
#pragma unroll 8
    for (int k = 0; k < FDIM; ++k) acc += (float)ef[(size_t)e * FDIM + k] * wsh[k];
    float s = 1.f / (1.f + __expf(-acc));
    float d = s - gt[e];
    err2 = d * d;
  }
  red[threadIdx.x] = err2;
  __syncthreads();
  for (int s = 128; s > 0; s >>= 1) {
    if (threadIdx.x < s) red[threadIdx.x] += red[threadIdx.x + s];
    __syncthreads();
  }
  if (threadIdx.x == 0) atomicAdd(out, red[0] * scale);
}

// ================= kernel: gather edge embeddings into subgraph rows ===============
__global__ __launch_bounds__(256) void gather_kernel(const bf16* __restrict__ ef,
                                                     const int* __restrict__ idx,
                                                     float* __restrict__ xg) {
  int i = blockIdx.x * 256 + threadIdx.x;
  if (i < GROWS * FDIM) {
    int r = i >> 6, k = i & 63;
    xg[i] = (float)ef[(size_t)idx[r] * FDIM + k];
  }
}

// ================= kernel: fused GGNN — fully wave-local, zero barriers ============
// Each wave owns 16 rows (1 or 2 whole ring subgraphs) and all 64 columns; the
// ring segment_sum is fused into the A-fragment load. All depth iterations run
// on wave-private LDS with no workgroup synchronization at all.
__global__ __launch_bounds__(256) void ggnn_kernel(
    const float* __restrict__ xg, const bf16* __restrict__ wmT,
    const float* __restrict__ bm, const bf16* __restrict__ wuT,
    const float* __restrict__ bu, int sgMask, int depth, float* __restrict__ pooled) {
  __shared__ float h[128 * FDIM];   // 32KB hidden state (f32 residual stream)
  __shared__ bf16  m[128 * FDIM];   // 16KB messages
  const int r0 = blockIdx.x * 128;
  const int wave = threadIdx.x >> 5;
  const int lane = threadIdx.x & 31;
  const int mt = wave * 16;                    // this wave's 16 rows

  // async-stage this wave's 16x64 f32 rows (4KB) into LDS
  {
    const char* g = (const char*)(xg + (size_t)(r0 + mt) * FDIM);
    char* l = (char*)&h[mt * FDIM];
#pragma unroll
    for (int o = 0; o < 4096; o += 512) async_copy_b128(l + o + lane * 16, g + o + lane * 16);
    async_wait0();
  }

  for (int d = 0; d < depth; ++d) {
    // m = LReLU(h @ wm + bm)     [16,64] x [64,64] — one A frag feeds 4 WMMAs
    {
      v8f c0 = {}, c1 = {}, c2 = {}, c3 = {};
#pragma unroll
      for (int kb = 0; kb < FDIM; kb += 32) {
        v16bf a = load_a_f32(h, FDIM, mt, kb);
        c0 = wmma_bf16(a, load_b(wmT, FDIM, 0,  kb), c0);
        c1 = wmma_bf16(a, load_b(wmT, FDIM, 16, kb), c1);
        c2 = wmma_bf16(a, load_b(wmT, FDIM, 32, kb), c2);
        c3 = wmma_bf16(a, load_b(wmT, FDIM, 48, kb), c3);
      }
      store_c_bf16(m, FDIM, mt, 0,  c0, bm, true);
      store_c_bf16(m, FDIM, mt, 16, c1, bm, true);
      store_c_bf16(m, FDIM, mt, 32, c2, bm, true);
      store_c_bf16(m, FDIM, mt, 48, c3, bm, true);
    }
    // h += LReLU((m[prev] + m[next]) @ wu + bu)  — neighbours stay in-tile
    {
      v8f c0 = {}, c1 = {}, c2 = {}, c3 = {};
#pragma unroll
      for (int kb = 0; kb < FDIM; kb += 32) {
        v16bf a = load_a_ringsum(m, FDIM, mt, kb, sgMask);
        c0 = wmma_bf16(a, load_b(wuT, FDIM, 0,  kb), c0);
        c1 = wmma_bf16(a, load_b(wuT, FDIM, 16, kb), c1);
        c2 = wmma_bf16(a, load_b(wuT, FDIM, 32, kb), c2);
        c3 = wmma_bf16(a, load_b(wuT, FDIM, 48, kb), c3);
      }
      int rOff = (lane & 16) ? 8 : 0;
      v8f* cs[4] = {&c0, &c1, &c2, &c3};
#pragma unroll
      for (int q = 0; q < 4; ++q) {
        int col = q * 16 + (lane & 15);
        float bb = bu[col];
#pragma unroll
        for (int r = 0; r < 8; ++r)
          h[(mt + r + rOff) * FDIM + col] += lrelu((*cs[q])[r] + bb);
      }
    }
  }

  // pooled mean per subgraph — still wave-local (sg <= 16, tile-aligned)
  int shift = __popc(sgMask);
  int nLocal = 16 >> shift;
  for (int i = lane; i < nLocal * FDIM; i += 32) {
    int s = i >> 6, k = i & 63;
    float acc = 0.f;
    for (int j = 0; j <= sgMask; ++j) acc += h[(mt + (s << shift) + j) * FDIM + k];
    pooled[((size_t)((r0 + mt) >> shift) + s) * FDIM + k] = acc / (float)(sgMask + 1);
  }
}

// ================= kernels: value head (BN stats / BN+act / WMMA GEMM / final) =====
__global__ __launch_bounds__(256) void colstat_kernel(const float* __restrict__ x, int rows,
                                                      int cols, float* __restrict__ sum,
                                                      float* __restrict__ sumsq) {
  int r0 = blockIdx.x * 64;
  int rend = r0 + 64 < rows ? r0 + 64 : rows;
  for (int c = threadIdx.x; c < cols; c += 256) {
    float s = 0.f, ss = 0.f;
    for (int r = r0; r < rend; ++r) {
      float v = x[(size_t)r * cols + c];
      s += v; ss += v * v;
    }
    atomicAdd(&sum[c], s);
    atomicAdd(&sumsq[c], ss);
  }
}

__global__ __launch_bounds__(256) void bnact_kernel(const float* __restrict__ x, int rows,
                                                    int cols, const float* __restrict__ sum,
                                                    const float* __restrict__ sumsq,
                                                    const float* __restrict__ g,
                                                    const float* __restrict__ b,
                                                    bf16* __restrict__ y) {
  int i = blockIdx.x * 256 + threadIdx.x;
  if (i >= rows * cols) return;
  int c = i % cols;
  float mean = sum[c] / rows;
  float var = sumsq[c] / rows - mean * mean;
  float v = (x[i] - mean) * rsqrtf(var + 1e-5f) * g[c] + b[c];
  y[i] = (bf16)lrelu(v);
}

__global__ __launch_bounds__(256) void gemm_kernel(const bf16* __restrict__ A, int K,
                                                   const bf16* __restrict__ Wt,
                                                   const float* __restrict__ bias, int Ncols,
                                                   float* __restrict__ out) {
  __shared__ bf16 As[32 * HLDIM];   // 32KB, K <= 512
  int r0 = blockIdx.x * 32;
  {
    // async-stage the contiguous [32][K] bf16 A tile into LDS
    const char* g = (const char*)(A + (size_t)r0 * K);
    char* l = (char*)As;
    int bytes = 32 * K * 2;
    for (int o = threadIdx.x * 16; o < bytes; o += 256 * 16) async_copy_b128(l + o, g + o);
    async_wait0();
  }
  __syncthreads();
  int wave = threadIdx.x >> 5;
  // one 64-col panel per wave across all 32 rows (Ncols == 512 -> 8 panels)
  for (int t = wave; t < (Ncols >> 6); t += 8) {
    int n0 = t * 64;
    CPanel2 p = gemm_panel2(As, K, Wt, K, n0);
    int lane = threadIdx.x & 31;
    int rOff = (lane & 16) ? 8 : 0;
#pragma unroll
    for (int q = 0; q < 4; ++q) {
      int col = n0 + q * 16 + (lane & 15);
      float bb = bias[col];
#pragma unroll
      for (int r = 0; r < 8; ++r) {
        out[(size_t)(r0 + r + rOff) * Ncols + col]      = p.c[q][r] + bb;
        out[(size_t)(r0 + 16 + r + rOff) * Ncols + col] = p.c[4 + q][r] + bb;
      }
    }
  }
}

__global__ __launch_bounds__(64) void vfinal_kernel(const float* __restrict__ pre2, int rows,
                                                    const float* __restrict__ sum2,
                                                    const float* __restrict__ sumsq2,
                                                    const float* __restrict__ g2,
                                                    const float* __restrict__ b2,
                                                    const float* __restrict__ w3,
                                                    const float* __restrict__ bb3,
                                                    float* __restrict__ out) {
  __shared__ float red[64];
  int row = blockIdx.x;
  float acc = 0.f;
  for (int c = threadIdx.x; c < HLDIM; c += 64) {
    float mean = sum2[c] / rows;
    float var = sumsq2[c] / rows - mean * mean;
    float v = (pre2[(size_t)row * HLDIM + c] - mean) * rsqrtf(var + 1e-5f) * g2[c] + b2[c];
    acc += lrelu(v) * w3[c];
  }
  red[threadIdx.x] = acc;
  __syncthreads();
  for (int s = 32; s > 0; s >>= 1) {
    if (threadIdx.x < s) red[threadIdx.x] += red[threadIdx.x + s];
    __syncthreads();
  }
  if (threadIdx.x == 0) out[row] = red[0] + bb3[0];
}

// =============================== host orchestration ================================
// Input order: setup_inputs() dict order; `params` flattens in JAX pytree order
// (dict keys sorted): g1[0..1]{bm,bu,wm,wu}, g2[0..1]{...}, q1{b0,b1,b2,bs,w0,w1,w2,ws},
// q2{...}, v1{b0,b1,b2,bb1,bb2,bb3,g0,g1,g2,w1,w2,w3}, v2{...}.
enum {
  IN_X = 0, IN_ACT, IN_ANG, IN_GT,
  IN_G1_0_BM, IN_G1_0_BU, IN_G1_0_WM, IN_G1_0_WU,
  IN_G1_1_BM, IN_G1_1_BU, IN_G1_1_WM, IN_G1_1_WU,
  IN_G2_0_BM, IN_G2_0_BU, IN_G2_0_WM, IN_G2_0_WU,
  IN_G2_1_BM, IN_G2_1_BU, IN_G2_1_WM, IN_G2_1_WU,
  IN_Q1_B0, IN_Q1_B1, IN_Q1_B2, IN_Q1_BS, IN_Q1_W0, IN_Q1_W1, IN_Q1_W2, IN_Q1_WS,
  IN_Q2_B0, IN_Q2_B1, IN_Q2_B2, IN_Q2_BS, IN_Q2_W0, IN_Q2_W1, IN_Q2_W2, IN_Q2_WS,
  IN_V1_B0, IN_V1_B1, IN_V1_B2, IN_V1_BB1, IN_V1_BB2, IN_V1_BB3,
  IN_V1_G0, IN_V1_G1, IN_V1_G2, IN_V1_W1, IN_V1_W2, IN_V1_W3,
  IN_V2_B0, IN_V2_B1, IN_V2_B2, IN_V2_BB1, IN_V2_BB2, IN_V2_BB3,
  IN_V2_G0, IN_V2_G1, IN_V2_G2, IN_V2_W1, IN_V2_W2, IN_V2_W3,
  IN_EI, IN_SUB0, IN_SUB1, IN_SEP0, IN_SEP1, IN_POST
};

extern "C" void kernel_launch(void* const* d_in, const int* in_sizes, int n_in,
                              void* d_out, int out_size, void* d_ws, size_t ws_size,
                              hipStream_t stream) {
  (void)in_sizes; (void)n_in; (void)out_size; (void)ws_size;
  auto F32 = [&](int i) { return (const float*)d_in[i]; };
  const int* ei   = (const int*)d_in[IN_EI];
  const int* sub0 = (const int*)d_in[IN_SUB0];
  const int* sub1 = (const int*)d_in[IN_SUB1];
  float* out = (float*)d_out;

  // ---- carve workspace ----
  uint8_t* base = (uint8_t*)d_ws;
  size_t off = 0;
  auto carve = [&](size_t bytes) -> void* {
    void* p = base + off;
    off += (bytes + 255) & ~(size_t)255;
    return p;
  };
  bf16* q1w0t = (bf16*)carve((size_t)HLDIM * K0PAD * 2);
  bf16* q1w1t = (bf16*)carve((size_t)HLDIM * HLDIM * 2);
  bf16* q1w2t = (bf16*)carve((size_t)FDIM * HLDIM * 2);
  bf16* q2w0t = (bf16*)carve((size_t)HLDIM * K0PAD * 2);
  bf16* q2w1t = (bf16*)carve((size_t)HLDIM * HLDIM * 2);
  bf16* q2w2t = (bf16*)carve((size_t)FDIM * HLDIM * 2);
  bf16* gT[8];
  for (int i = 0; i < 8; ++i) gT[i] = (bf16*)carve((size_t)FDIM * FDIM * 2);
  bf16* v1w1t = (bf16*)carve((size_t)HLDIM * FDIM * 2);
  bf16* v1w2t = (bf16*)carve((size_t)HLDIM * HLDIM * 2);
  bf16* v2w1t = (bf16*)carve((size_t)HLDIM * FDIM * 2);
  bf16* v2w2t = (bf16*)carve((size_t)HLDIM * HLDIM * 2);
  bf16* ef1   = (bf16*)carve((size_t)NEDGES * FDIM * 2);
  bf16* ef2   = (bf16*)carve((size_t)NEDGES * FDIM * 2);
  float* xg     = (float*)carve((size_t)GROWS * FDIM * 4);
  float* pooled = (float*)carve((size_t)NSG0 * FDIM * 4);
  float* pre1   = (float*)carve((size_t)NSG0 * HLDIM * 4);
  float* pre2   = (float*)carve((size_t)NSG0 * HLDIM * 4);
  bf16*  actbf  = (bf16*)carve((size_t)NSG0 * HLDIM * 2);
  float* stats  = (float*)carve((size_t)(2 * FDIM + 4 * HLDIM) * 4);
  float *sum0 = stats, *sq0 = stats + FDIM;
  float *sum1 = stats + 2 * FDIM, *sq1 = sum1 + HLDIM;
  float *sum2 = sq1 + HLDIM, *sq2 = sum2 + HLDIM;
  const size_t statsBytes = (size_t)(2 * FDIM + 4 * HLDIM) * 4;

  // ---- weight prep (f32 -> bf16, transposed [N][Kpad]) ----
  auto prep = [&](int in, int K, int Nc, int Kpad, bf16* dst) {
    wprep_kernel<<<(Nc * Kpad + 255) / 256, 256, 0, stream>>>(F32(in), K, Nc, Kpad, dst);
  };
  prep(IN_Q1_W0, 132, HLDIM, K0PAD, q1w0t);
  prep(IN_Q1_W1, HLDIM, HLDIM, HLDIM, q1w1t);
  prep(IN_Q1_W2, HLDIM, FDIM, HLDIM, q1w2t);
  prep(IN_Q2_W0, 132, HLDIM, K0PAD, q2w0t);
  prep(IN_Q2_W1, HLDIM, HLDIM, HLDIM, q2w1t);
  prep(IN_Q2_W2, HLDIM, FDIM, HLDIM, q2w2t);
  int gW[8] = {IN_G1_0_WM, IN_G1_0_WU, IN_G1_1_WM, IN_G1_1_WU,
               IN_G2_0_WM, IN_G2_0_WU, IN_G2_1_WM, IN_G2_1_WU};
  for (int i = 0; i < 8; ++i) prep(gW[i], FDIM, FDIM, FDIM, gT[i]);
  prep(IN_V1_W1, FDIM, HLDIM, FDIM, v1w1t);
  prep(IN_V1_W2, HLDIM, HLDIM, HLDIM, v1w2t);
  prep(IN_V2_W1, FDIM, HLDIM, FDIM, v2w1t);
  prep(IN_V2_W2, HLDIM, HLDIM, HLDIM, v2w2t);

  // ---- fused edge MLPs (dominant FLOPs) ----
  qgnn_kernel<<<NEDGES / 32, 256, 0, stream>>>(F32(IN_X), F32(IN_ACT), F32(IN_ANG), ei,
                                               q1w0t, F32(IN_Q1_B0), q1w1t, F32(IN_Q1_B1),
                                               q1w2t, F32(IN_Q1_B2), ef1);
  qgnn_kernel<<<NEDGES / 32, 256, 0, stream>>>(F32(IN_X), F32(IN_ACT), F32(IN_ANG), ei,
                                               q2w0t, F32(IN_Q2_B0), q2w1t, F32(IN_Q2_B1),
                                               q2w2t, F32(IN_Q2_B2), ef2);

  // ---- side loss: side = (mse1 + mse2) / 4 ----
  const int SIDE_OFF = 2 * (NSG0 + NSG1);   // 49152
  hipMemsetAsync(out + SIDE_OFF, 0, sizeof(float), stream);
  const float scale = 1.f / (4.f * (float)NEDGES);
  side_kernel<<<(NEDGES + 255) / 256, 256, 0, stream>>>(ef1, F32(IN_Q1_WS), F32(IN_Q1_BS),
                                                        F32(IN_GT), scale, out + SIDE_OFF);
  side_kernel<<<(NEDGES + 255) / 256, 256, 0, stream>>>(ef2, F32(IN_Q2_WS), F32(IN_Q2_BS),
                                                        F32(IN_GT), scale, out + SIDE_OFF);

  // ---- one (subgraph-set, net) pass: gather -> fused GGNN -> value head ----
  auto run_pass = [&](const bf16* ef, const int* subIdx, int sgMask, int depth,
                      const bf16* wmT, const float* bm, const bf16* wuT, const float* bu,
                      int nsg,
                      const float* vg0, const float* vb0, const bf16* vw1t, const float* vbb1,
                      const float* vg1, const float* vb1, const bf16* vw2t, const float* vbb2,
                      const float* vg2, const float* vb2, const float* vw3, const float* vbb3,
                      float* outSlice) {
    gather_kernel<<<(GROWS * FDIM + 255) / 256, 256, 0, stream>>>(ef, subIdx, xg);
    ggnn_kernel<<<GROWS / 128, 256, 0, stream>>>(xg, wmT, bm, wuT, bu, sgMask, depth, pooled);
    hipMemsetAsync(stats, 0, statsBytes, stream);
    colstat_kernel<<<(nsg + 63) / 64, 256, 0, stream>>>(pooled, nsg, FDIM, sum0, sq0);
    bnact_kernel<<<(nsg * FDIM + 255) / 256, 256, 0, stream>>>(pooled, nsg, FDIM, sum0, sq0,
                                                               vg0, vb0, actbf);
    gemm_kernel<<<nsg / 32, 256, 0, stream>>>(actbf, FDIM, vw1t, vbb1, HLDIM, pre1);
    colstat_kernel<<<(nsg + 63) / 64, 256, 0, stream>>>(pre1, nsg, HLDIM, sum1, sq1);
    bnact_kernel<<<(nsg * HLDIM + 255) / 256, 256, 0, stream>>>(pre1, nsg, HLDIM, sum1, sq1,
                                                                vg1, vb1, actbf);
    gemm_kernel<<<nsg / 32, 256, 0, stream>>>(actbf, HLDIM, vw2t, vbb2, HLDIM, pre2);
    colstat_kernel<<<(nsg + 63) / 64, 256, 0, stream>>>(pre2, nsg, HLDIM, sum2, sq2);
    vfinal_kernel<<<nsg, 64, 0, stream>>>(pre2, nsg, sum2, sq2, vg2, vb2, vw3, vbb3, outSlice);
  };

  // outs1 (net1 / value v1): set0 then set1; outs2 (net2 / v2): set0 then set1
  run_pass(ef1, sub0, 7, 4, gT[0], F32(IN_G1_0_BM), gT[1], F32(IN_G1_0_BU), NSG0,
           F32(IN_V1_G0), F32(IN_V1_B0), v1w1t, F32(IN_V1_BB1),
           F32(IN_V1_G1), F32(IN_V1_B1), v1w2t, F32(IN_V1_BB2),
           F32(IN_V1_G2), F32(IN_V1_B2), F32(IN_V1_W3), F32(IN_V1_BB3), out + 0);
  run_pass(ef1, sub1, 15, 8, gT[2], F32(IN_G1_1_BM), gT[3], F32(IN_G1_1_BU), NSG1,
           F32(IN_V1_G0), F32(IN_V1_B0), v1w1t, F32(IN_V1_BB1),
           F32(IN_V1_G1), F32(IN_V1_B1), v1w2t, F32(IN_V1_BB2),
           F32(IN_V1_G2), F32(IN_V1_B2), F32(IN_V1_W3), F32(IN_V1_BB3), out + NSG0);
  run_pass(ef2, sub0, 7, 4, gT[4], F32(IN_G2_0_BM), gT[5], F32(IN_G2_0_BU), NSG0,
           F32(IN_V2_G0), F32(IN_V2_B0), v2w1t, F32(IN_V2_BB1),
           F32(IN_V2_G1), F32(IN_V2_B1), v2w2t, F32(IN_V2_BB2),
           F32(IN_V2_G2), F32(IN_V2_B2), F32(IN_V2_W3), F32(IN_V2_BB3), out + NSG0 + NSG1);
  run_pass(ef2, sub1, 15, 8, gT[6], F32(IN_G2_1_BM), gT[7], F32(IN_G2_1_BU), NSG1,
           F32(IN_V2_G0), F32(IN_V2_B0), v2w1t, F32(IN_V2_BB1),
           F32(IN_V2_G1), F32(IN_V2_B1), v2w2t, F32(IN_V2_BB2),
           F32(IN_V2_G2), F32(IN_V2_B2), F32(IN_V2_W3), F32(IN_V2_BB3),
           out + 2 * NSG0 + NSG1);
}
